// SelfAttention_1778116460733
// MI455X (gfx1250) — matrix operand order repaired and checked by reading, not compiled, not run on hardware
//
#include <hip/hip_runtime.h>

#define SEQ    2048
#define DMODEL 2048
#define NHEAD  16
#define HDIM   128

typedef __attribute__((ext_vector_type(16))) __bf16 v16bf;
typedef __attribute__((ext_vector_type(8)))  float  v8f;

union FragU {
    unsigned short s[16];
    unsigned int   u[8];
    uint4 u4[2];
    v16bf v;
};

// bf16 via truncation (1 shift; below bf16 rounding noise for this net)
__device__ inline unsigned short f2bf(float f) {
    return (unsigned short)(__float_as_uint(f) >> 16);
}
// pack two fp32 -> two bf16 (lo,hi) in ONE v_perm_b32
__device__ inline unsigned int pkbf(float lo, float hi) {
    return __builtin_amdgcn_perm(__float_as_uint(hi), __float_as_uint(lo), 0x07060302u);
}

// ---------------------------------------------------------------------------
// Kernel 1: C[M,N] = A[M,K] * W[N,K]^T + bias[N]   (fp32 in/out, bf16 WMMA)
// Block tile 128x128, 8 waves in 4x2 grid, each wave 32x64 (2x4 WMMA tiles).
// ---------------------------------------------------------------------------
__global__ void __launch_bounds__(256)
gemm_wmma_kernel(const float* __restrict__ A, const float* __restrict__ W,
                 const float* __restrict__ bias, float* __restrict__ C,
                 int M, int N, int Ktot)
{
    __shared__ __align__(16) unsigned short As[128][40];  // [m][k], 80B row stride
    __shared__ __align__(16) unsigned short Ws[128][40];  // [n][k]

    const int tid  = threadIdx.x;
    const int wave = tid >> 5;
    const int lane = tid & 31;
    const int hf   = lane >> 4;   // lane half
    const int l16  = lane & 15;
    const int wm   = wave >> 1;   // 0..3 (rows of 32)
    const int wn   = wave & 1;    // 0..1 (cols of 64)
    const int blockM = blockIdx.y * 128;
    const int blockN = blockIdx.x * 128;

    const int fq = tid & 7;       // float4 slot within a 32-wide k row
    const int rr = tid >> 3;      // base row (32 rows per pass)

    v8f zero = {0.f,0.f,0.f,0.f,0.f,0.f,0.f,0.f};
    v8f acc[2][4];
    #pragma unroll
    for (int mi = 0; mi < 2; ++mi)
        #pragma unroll
        for (int ni = 0; ni < 4; ++ni) acc[mi][ni] = zero;

    for (int k0 = 0; k0 < Ktot; k0 += 32) {
        // Batch ALL staging loads first (distinct regs -> pipelined loadcnt drain)
        float4 fa[4], fw[4];
        #pragma unroll
        for (int p = 0; p < 4; ++p)
            fa[p] = *(const float4*)&A[(size_t)(blockM + rr + p * 32) * Ktot + k0 + fq * 4];
        #pragma unroll
        for (int p = 0; p < 4; ++p)
            fw[p] = *(const float4*)&W[(size_t)(blockN + rr + p * 32) * Ktot + k0 + fq * 4];

        __syncthreads();
        #pragma unroll
        for (int p = 0; p < 4; ++p) {
            uint2 u;
            u.x = pkbf(fa[p].x, fa[p].y);
            u.y = pkbf(fa[p].z, fa[p].w);
            *(uint2*)&As[rr + p * 32][fq * 4] = u;
        }
        #pragma unroll
        for (int p = 0; p < 4; ++p) {
            uint2 u;
            u.x = pkbf(fw[p].x, fw[p].y);
            u.y = pkbf(fw[p].z, fw[p].w);
            *(uint2*)&Ws[rr + p * 32][fq * 4] = u;
        }
        // Prefetch next K chunk while WMMAs run (global_prefetch_b8)
        if (k0 + 32 < Ktot) {
            __builtin_prefetch(&A[(size_t)(blockM + rr) * Ktot + k0 + 32 + fq * 4], 0, 3);
            __builtin_prefetch(&W[(size_t)(blockN + rr) * Ktot + k0 + 32 + fq * 4], 0, 3);
        }
        __syncthreads();

        FragU af[2], bfm[4];
        #pragma unroll
        for (int mi = 0; mi < 2; ++mi) {
            // A layout: lane<16 -> K 0..7 / 16..23 ; lane>=16 -> K 8..15 / 24..31
            int row = wm * 32 + mi * 16 + l16;
            int k1  = hf * 8;
            af[mi].u4[0] = *(const uint4*)&As[row][k1];
            af[mi].u4[1] = *(const uint4*)&As[row][k1 + 16];
        }
        #pragma unroll
        for (int ni = 0; ni < 4; ++ni) {
            // B layout: lane<16 -> K 0..7 / 16..23 ; lane>=16 -> K 8..15 / 24..31
            int col = wn * 64 + ni * 16 + l16;
            int kb  = hf * 8;
            bfm[ni].u4[0] = *(const uint4*)&Ws[col][kb];
            bfm[ni].u4[1] = *(const uint4*)&Ws[col][kb + 16];
        }
        #pragma unroll
        for (int mi = 0; mi < 2; ++mi)
            #pragma unroll
            for (int ni = 0; ni < 4; ++ni)
                acc[mi][ni] = __builtin_amdgcn_wmma_f32_16x16x32_bf16(
                    false, af[mi].v, false, bfm[ni].v,
                    (short)0, acc[mi][ni], false, false);
    }

    // Store: C tile layout -> row = hf*8 + r, col = l16
    #pragma unroll
    for (int mi = 0; mi < 2; ++mi)
        #pragma unroll
        for (int ni = 0; ni < 4; ++ni) {
            int col = blockN + wn * 64 + ni * 16 + l16;
            float b = bias[col];
            #pragma unroll
            for (int r = 0; r < 8; ++r) {
                int row = blockM + wm * 32 + mi * 16 + hf * 8 + r;
                C[(size_t)row * N + col] = acc[mi][ni][r] + b;
            }
        }
}

// ---------------------------------------------------------------------------
// Kernel 2: fused RMSNorm (over full D) + RoPE, in place. One block per row.
// ---------------------------------------------------------------------------
__global__ void __launch_bounds__(256)
rmsnorm_rope_kernel(float* __restrict__ buf, const float* __restrict__ g,
                    const float* __restrict__ cosb, const float* __restrict__ sinb)
{
    __shared__ float red[8];
    const int s   = blockIdx.x;
    const int tid = threadIdx.x;
    float* row = buf + (size_t)s * DMODEL;

    float x[8];
    float ss = 0.f;
    #pragma unroll
    for (int i = 0; i < 8; ++i) { x[i] = row[tid * 8 + i]; ss += x[i] * x[i]; }
    #pragma unroll
    for (int off = 16; off; off >>= 1) ss += __shfl_xor(ss, off, 32);
    if ((tid & 31) == 0) red[tid >> 5] = ss;
    __syncthreads();
    float tot = 0.f;
    #pragma unroll
    for (int w = 0; w < 8; ++w) tot += red[w];
    float rs = rsqrtf(tot / (float)DMODEL + 1e-6f);

    const float* cr = cosb + (size_t)s * (HDIM / 2);
    const float* sr = sinb + (size_t)s * (HDIM / 2);
    #pragma unroll
    for (int i = 0; i < 8; i += 2) {
        int d = tid * 8 + i;
        float a = x[i]     * rs * g[d];
        float b = x[i + 1] * rs * g[d + 1];
        int p = (d & (HDIM - 1)) >> 1;
        float c = cr[p], si = sr[p];
        row[d]     = a * c - b * si;
        row[d + 1] = a * si + b * c;
    }
}

// ---------------------------------------------------------------------------
// Kernel 3: flash attention. grid = (S/128 q-tiles, H heads), block = 256.
// Each wave owns 16 query rows; block iterates kv in tiles of 32.
// ---------------------------------------------------------------------------
__global__ void __launch_bounds__(256)
attn_wmma_kernel(const float* __restrict__ Q, const float* __restrict__ K,
                 const float* __restrict__ V, float* __restrict__ O)
{
    __shared__ __align__(16) unsigned short Kt[32][136];    // [kv][d], 272B stride
    __shared__ __align__(16) unsigned short Vt[128][40];    // [d][kv] (transposed)
    __shared__ __align__(16) unsigned short Pw[8][16][40];  // per-wave P staging

    const int tid   = threadIdx.x;
    const int wave  = tid >> 5;
    const int lane  = tid & 31;
    const int hf    = lane >> 4;
    const int l16   = lane & 15;
    const int head  = blockIdx.y;
    const int qbase = blockIdx.x * 128 + wave * 16;
    const int dhead = head * HDIM;
    const float scale = 0.08838834764831845f;  // 1/sqrt(128)

    // Q fragments (A layout) straight from global: float4 loads + perm packing
    FragU qf[4];
    {
        const float* qrow = Q + (size_t)(qbase + l16) * DMODEL + dhead;
        #pragma unroll
        for (int c = 0; c < 4; ++c) {
            int k1 = c * 32 + hf * 8;
            int k2 = k1 + 16;
            float4 a0 = *(const float4*)&qrow[k1];
            float4 a1 = *(const float4*)&qrow[k1 + 4];
            float4 b0 = *(const float4*)&qrow[k2];
            float4 b1 = *(const float4*)&qrow[k2 + 4];
            qf[c].u[0] = pkbf(a0.x, a0.y);
            qf[c].u[1] = pkbf(a0.z, a0.w);
            qf[c].u[2] = pkbf(a1.x, a1.y);
            qf[c].u[3] = pkbf(a1.z, a1.w);
            qf[c].u[4] = pkbf(b0.x, b0.y);
            qf[c].u[5] = pkbf(b0.z, b0.w);
            qf[c].u[6] = pkbf(b1.x, b1.y);
            qf[c].u[7] = pkbf(b1.z, b1.w);
        }
    }

    v8f zero = {0.f,0.f,0.f,0.f,0.f,0.f,0.f,0.f};
    v8f acc[8];
    #pragma unroll
    for (int t = 0; t < 8; ++t) acc[t] = zero;
    float mrow[8], lrow[8];
    #pragma unroll
    for (int r = 0; r < 8; ++r) { mrow[r] = -1e30f; lrow[r] = 0.f; }

    const int fq = tid & 31;   // float4 slot within 128-wide d row
    const int rr = tid >> 5;   // kv row base (8 per pass)

    for (int kv0 = 0; kv0 < SEQ; kv0 += 32) {
        // Batch ALL staging loads first (pipelined loadcnt drain)
        float4 fk[4], fv[4];
        #pragma unroll
        for (int p = 0; p < 4; ++p)
            fk[p] = *(const float4*)&K[(size_t)(kv0 + rr + p * 8) * DMODEL + dhead + fq * 4];
        #pragma unroll
        for (int p = 0; p < 4; ++p)
            fv[p] = *(const float4*)&V[(size_t)(kv0 + rr + p * 8) * DMODEL + dhead + fq * 4];

        __syncthreads();
        // Stage K tile [32 kv][128 d]: perm-pack + b64 stores
        #pragma unroll
        for (int p = 0; p < 4; ++p) {
            uint2 u;
            u.x = pkbf(fk[p].x, fk[p].y);
            u.y = pkbf(fk[p].z, fk[p].w);
            *(uint2*)&Kt[rr + p * 8][fq * 4] = u;
        }
        // Stage V tile transposed: scattered b16 stores
        #pragma unroll
        for (int p = 0; p < 4; ++p) {
            int r = rr + p * 8;
            int d = fq * 4;
            Vt[d][r]     = f2bf(fv[p].x);
            Vt[d + 1][r] = f2bf(fv[p].y);
            Vt[d + 2][r] = f2bf(fv[p].z);
            Vt[d + 3][r] = f2bf(fv[p].w);
        }
        if (kv0 + 32 < SEQ) {
            __builtin_prefetch(&K[(size_t)(kv0 + 32 + rr) * DMODEL + dhead + fq * 4], 0, 3);
            __builtin_prefetch(&V[(size_t)(kv0 + 32 + rr) * DMODEL + dhead + fq * 4], 0, 3);
        }
        __syncthreads();

        // Scores: two 16x16 C tiles (kv cols 0-15 and 16-31), K-dim = 128
        v8f s0 = zero, s1 = zero;
        #pragma unroll
        for (int c = 0; c < 4; ++c) {
            FragU b0, b1;
            int dbase = c * 32 + hf * 8;
            b0.u4[0] = *(const uint4*)&Kt[l16][dbase];
            b0.u4[1] = *(const uint4*)&Kt[l16][dbase + 16];
            b1.u4[0] = *(const uint4*)&Kt[l16 + 16][dbase];
            b1.u4[1] = *(const uint4*)&Kt[l16 + 16][dbase + 16];
            s0 = __builtin_amdgcn_wmma_f32_16x16x32_bf16(false, qf[c].v, false, b0.v,
                                                         (short)0, s0, false, false);
            s1 = __builtin_amdgcn_wmma_f32_16x16x32_bf16(false, qf[c].v, false, b1.v,
                                                         (short)0, s1, false, false);
        }

        // Online softmax per row (rows split across lane halves; cols across 16 lanes)
        float corr[8];
        #pragma unroll
        for (int r = 0; r < 8; ++r) {
            float v = fmaxf(s0[r], s1[r]);
            #pragma unroll
            for (int off = 8; off; off >>= 1) v = fmaxf(v, __shfl_xor(v, off, 16));
            float mnew = fmaxf(mrow[r], v * scale);
            corr[r] = __expf(mrow[r] - mnew);
            mrow[r] = mnew;
            float p0 = __expf(s0[r] * scale - mnew);
            float p1 = __expf(s1[r] * scale - mnew);
            s0[r] = p0; s1[r] = p1;
            float sum = p0 + p1;
            #pragma unroll
            for (int off = 8; off; off >>= 1) sum += __shfl_xor(sum, off, 16);
            lrow[r] = lrow[r] * corr[r] + sum;
        }
        #pragma unroll
        for (int t = 0; t < 8; ++t)
            #pragma unroll
            for (int r = 0; r < 8; ++r) acc[t][r] *= corr[r];

        // C-layout -> A-layout conversion for P via per-wave LDS patch
        #pragma unroll
        for (int r = 0; r < 8; ++r) {
            int row = hf * 8 + r;
            Pw[wave][row][l16]      = f2bf(s0[r]);
            Pw[wave][row][l16 + 16] = f2bf(s1[r]);
        }
        FragU pf;
        {
            int k1 = hf * 8;
            pf.u4[0] = *(const uint4*)&Pw[wave][l16][k1];
            pf.u4[1] = *(const uint4*)&Pw[wave][l16][k1 + 16];
        }

        // O += P(16x32) * V(32x128) over 8 N-subtiles
        #pragma unroll
        for (int n = 0; n < 8; ++n) {
            FragU bv;
            int dcol = n * 16 + l16;
            int kb   = hf * 8;
            bv.u4[0] = *(const uint4*)&Vt[dcol][kb];
            bv.u4[1] = *(const uint4*)&Vt[dcol][kb + 16];
            acc[n] = __builtin_amdgcn_wmma_f32_16x16x32_bf16(false, pf.v, false, bv.v,
                                                             (short)0, acc[n], false, false);
        }
    }

    // Epilogue: divide by l, write back [s, head*128 + d]
    #pragma unroll
    for (int n = 0; n < 8; ++n) {
        int col = dhead + n * 16 + l16;
        #pragma unroll
        for (int r = 0; r < 8; ++r) {
            int row = qbase + hf * 8 + r;
            O[(size_t)row * DMODEL + col] = acc[n][r] / lrow[r];
        }
    }
}

// ---------------------------------------------------------------------------
extern "C" void kernel_launch(void* const* d_in, const int* in_sizes, int n_in,
                              void* d_out, int out_size, void* d_ws, size_t ws_size,
                              hipStream_t stream) {
    const float* x   = (const float*)d_in[0];
    const float* wq  = (const float*)d_in[1];
    const float* bq  = (const float*)d_in[2];
    const float* wk  = (const float*)d_in[3];
    const float* bk  = (const float*)d_in[4];
    const float* wv  = (const float*)d_in[5];
    const float* bv  = (const float*)d_in[6];
    const float* wo  = (const float*)d_in[7];
    const float* bo  = (const float*)d_in[8];
    const float* gq  = (const float*)d_in[9];
    const float* gk  = (const float*)d_in[10];
    const float* fc  = (const float*)d_in[11];
    const float* fs  = (const float*)d_in[12];
    float* out = (float*)d_out;

    const size_t mat = (size_t)SEQ * DMODEL;
    float* q    = (float*)d_ws;
    float* k    = q + mat;
    float* v    = k + mat;
    float* attn = v + mat;

    dim3 gg(DMODEL / 128, SEQ / 128);
    gemm_wmma_kernel<<<gg, 256, 0, stream>>>(x, wq, bq, q, SEQ, DMODEL, DMODEL);
    gemm_wmma_kernel<<<gg, 256, 0, stream>>>(x, wk, bk, k, SEQ, DMODEL, DMODEL);
    gemm_wmma_kernel<<<gg, 256, 0, stream>>>(x, wv, bv, v, SEQ, DMODEL, DMODEL);

    rmsnorm_rope_kernel<<<SEQ, 256, 0, stream>>>(q, gq, fc, fs);
    rmsnorm_rope_kernel<<<SEQ, 256, 0, stream>>>(k, gk, fc, fs);

    attn_wmma_kernel<<<dim3(SEQ / 128, NHEAD), 256, 0, stream>>>(q, k, v, attn);

    gemm_wmma_kernel<<<gg, 256, 0, stream>>>(attn, wo, bo, out, SEQ, DMODEL, DMODEL);
}
